// MultiHeadAttention_52158082843478
// MI455X (gfx1250) — matrix operand order repaired and checked by reading, not compiled, not run on hardware
//
#include <hip/hip_runtime.h>

typedef __attribute__((ext_vector_type(16))) _Float16 v16h;
typedef __attribute__((ext_vector_type(8)))  _Float16 v8h;
typedef __attribute__((ext_vector_type(8)))  float    v8f;
typedef __attribute__((ext_vector_type(4)))  int      v4i;

#define BATCH   4
#define SEQ     2048
#define CDIM    1024
#define NHEAD   16
#define DHEAD   64

// ---------------------------------------------------------------------------
// Async global->LDS staging (CDNA5 GLOBAL_LOAD_ASYNC_TO_LDS_B128, ASYNCcnt),
// guarded so the kernel still compiles on toolchains without the builtin.
// Signature (from compiler diagnostic): (global v4i*, shared v4i*, imm, imm).
// ---------------------------------------------------------------------------
#if defined(__has_builtin)
#  if __has_builtin(__builtin_amdgcn_global_load_async_to_lds_b128)
#    define ASYNC_LDS 1
#  endif
#endif
#ifndef ASYNC_LDS
#  define ASYNC_LDS 0
#endif

typedef __attribute__((address_space(1))) v4i gv4i;
typedef __attribute__((address_space(3))) v4i lv4i;

__device__ inline void copy16_g2l(const _Float16* g, _Float16* l) {
#if ASYNC_LDS
    __builtin_amdgcn_global_load_async_to_lds_b128(
        (gv4i*)(void*)g, (lv4i*)(void*)l, 0, 0);
#else
    *(uint4*)l = *(const uint4*)g;
#endif
}

__device__ inline void wait_stage() {
#if ASYNC_LDS
#  if __has_builtin(__builtin_amdgcn_s_wait_asynccnt)
    __builtin_amdgcn_s_wait_asynccnt(0);
#  else
    asm volatile("s_wait_asynccnt 0x0" ::: "memory");
#  endif
#endif
}

// ---------------------------------------------------------------------------
// Fragment loaders (CDNA5 wave32 WMMA layouts, ISA 7.12.2)
// A 16x32 f16: lane<16 -> row=lane, K = {0..7, 16..23}; lane>=16 -> K = {8..15, 24..31}
// B 32x16 f16 from [N][K] row-major tile: lane holds N=lane&15,
//   K = same as A: lane<16 -> {0..7, 16..23}; lane>=16 -> {8..15, 24..31}
// C/D 16x16 f32: reg r, lane -> row = r + 8*(lane>=16), col = lane&15
// ---------------------------------------------------------------------------
__device__ inline v16h lds_frag_a(const _Float16* tile, int stride, int lane, int k0) {
    int row  = lane & 15;
    int kOff = (lane < 16) ? 0 : 8;
    const _Float16* p = tile + row * stride + k0 + kOff;
    union { v16h v; v8h h[2]; } u;
    u.h[0] = *(const v8h*)p;
    u.h[1] = *(const v8h*)(p + 16);
    return u.v;
}
__device__ inline v16h lds_frag_b(const _Float16* tile, int stride, int lane, int k0) {
    int row  = lane & 15;
    int kOff = (lane < 16) ? 0 : 8;
    const _Float16* p = tile + row * stride + k0 + kOff;
    union { v16h v; v8h h[2]; } u;
    u.h[0] = *(const v8h*)p;
    u.h[1] = *(const v8h*)(p + 16);
    return u.v;
}

// ---------------------------------------------------------------------------
// Prep: fp32 -> fp16 convert / transpose-convert
// ---------------------------------------------------------------------------
__global__ void cvt_f32_f16(const float* __restrict__ in, _Float16* __restrict__ out, int n) {
    int i = blockIdx.x * blockDim.x + threadIdx.x;
    if (i < n) out[i] = (_Float16)in[i];
}

// out[b][c][r] = in[b][r][c]   (batch of rows x cols transposes, with f32->f16)
__global__ void transpose_cvt(const float* __restrict__ in, _Float16* __restrict__ out,
                              int batch, int rows, int cols) {
    int i = blockIdx.x * blockDim.x + threadIdx.x;
    int total = batch * rows * cols;
    if (i < total) {
        int b  = i / (rows * cols);
        int rc = i % (rows * cols);
        int r  = rc / cols;
        int c  = rc % cols;
        out[(size_t)b * rows * cols + (size_t)c * rows + r] = (_Float16)in[i];
    }
}

// ---------------------------------------------------------------------------
// QKV projection: per (b,h,sel) 128x64 tile of x[T,C] @ W[C,D] via WT[D,C].
// 4 waves; each wave owns a 32-row strip (2 A frags x 4 B frags = 8 wmma/chunk).
// Double-buffered LDS, async global->LDS fill overlapped with WMMA.
// q is pre-scaled by 1024^-0.05 = 2^-0.5.
// ---------------------------------------------------------------------------
__global__ __launch_bounds__(128)
void qkv_gemm(const _Float16* __restrict__ xh,     // [B*T, C]
              const _Float16* __restrict__ WqT,    // [H, D, C]
              const _Float16* __restrict__ WkT,
              const _Float16* __restrict__ WvT,
              _Float16* __restrict__ qo,           // [B*H, T, D]
              _Float16* __restrict__ ko,
              _Float16* __restrict__ vo) {
    int tileT = blockIdx.x;                 // T/128
    int bh    = blockIdx.y;                 // B*H
    int sel   = blockIdx.z;                 // 0=q 1=k 2=v
    int b = bh / NHEAD, h = bh % NHEAD;

    const _Float16* W = (sel == 0 ? WqT : sel == 1 ? WkT : WvT) + (size_t)h * DHEAD * CDIM;
    _Float16* out = (sel == 0 ? qo : sel == 1 ? ko : vo) + (size_t)bh * SEQ * DHEAD;
    float scale = (sel == 0) ? 0.70710678118654752f : 1.0f;
    const _Float16* Arow = xh + ((size_t)b * SEQ + (size_t)tileT * 128) * CDIM;

    __shared__ __align__(16) _Float16 As[2][128 * 32];
    __shared__ __align__(16) _Float16 Bs[2][64 * 32];

    int tid = threadIdx.x, wave = tid >> 5, lane = tid & 31;
    v8f acc[2][4] = {};

    auto stage = [&](int buf, int kc) {
        for (int i = tid; i < 512; i += 128) {           // 128 rows x 4 segs of 8 halves
            int r = i >> 2, seg = i & 3;
            copy16_g2l(Arow + (size_t)r * CDIM + kc + seg * 8, &As[buf][r * 32 + seg * 8]);
        }
        for (int i = tid; i < 256; i += 128) {           // 64 rows x 4 segs
            int r = i >> 2, seg = i & 3;
            copy16_g2l(W + (size_t)r * CDIM + kc + seg * 8, &Bs[buf][r * 32 + seg * 8]);
        }
    };

    stage(0, 0);
    for (int kc = 0, it = 0; kc < CDIM; kc += 32, ++it) {
        int cur = it & 1;
        wait_stage();
        __syncthreads();                                  // publish buffer `cur`
        if (kc + 32 < CDIM) stage(cur ^ 1, kc + 32);      // overlap next fill with WMMA
        v16h a0 = lds_frag_a(&As[cur][(wave * 32 + 0)  * 32], 32, lane, 0);
        v16h a1 = lds_frag_a(&As[cur][(wave * 32 + 16) * 32], 32, lane, 0);
#pragma unroll
        for (int n = 0; n < 4; ++n) {
            v16h bf = lds_frag_b(&Bs[cur][n * 16 * 32], 32, lane, 0);
            acc[0][n] = __builtin_amdgcn_wmma_f32_16x16x32_f16(false, a0, false, bf,
                                                               (short)0, acc[0][n], false, false);
            acc[1][n] = __builtin_amdgcn_wmma_f32_16x16x32_f16(false, a1, false, bf,
                                                               (short)0, acc[1][n], false, false);
        }
    }

    int mbase = (lane < 16) ? 0 : 8;
    int col = lane & 15;
#pragma unroll
    for (int sub = 0; sub < 2; ++sub)
#pragma unroll
        for (int n = 0; n < 4; ++n)
#pragma unroll
            for (int r = 0; r < 8; ++r) {
                int t = tileT * 128 + wave * 32 + sub * 16 + mbase + r;
                out[(size_t)t * DHEAD + n * 16 + col] = (_Float16)(acc[sub][n][r] * scale);
            }
}

// ---------------------------------------------------------------------------
// Flash attention per (b,h) and 64-row q tile. Online softmax, causal.
// ---------------------------------------------------------------------------
__global__ __launch_bounds__(128)
void attn_kernel(const _Float16* __restrict__ qg,  // [B*H, T, D] (scaled)
                 const _Float16* __restrict__ kg,
                 const _Float16* __restrict__ vg,
                 _Float16* __restrict__ Y) {        // [B*T, H*D]
    int tq = blockIdx.x;                  // T/64
    int bh = blockIdx.y;
    int b = bh / NHEAD, h = bh % NHEAD;
    const _Float16* qp = qg + (size_t)bh * SEQ * DHEAD;
    const _Float16* kp = kg + (size_t)bh * SEQ * DHEAD;
    const _Float16* vp = vg + (size_t)bh * SEQ * DHEAD;

    __shared__ __align__(16) _Float16 Ks[64 * 64];       // [s][d]
    __shared__ __align__(16) _Float16 Vs[64 * 64];       // transposed: [d][s]
    __shared__ __align__(16) _Float16 Ps[4 * 16 * 64];   // per-wave P strip [16][64]

    int tid = threadIdx.x, wave = tid >> 5, lane = tid & 31;
    int t0 = tq * 64;
    int mbase = (lane < 16) ? 0 : 8;
    int col = lane & 15;

    // Stage q tile (borrow Ks), grab per-wave A fragments into registers.
    for (int i = tid; i < 512; i += 128) {
        int r = i >> 3, seg = i & 7;
        *(uint4*)(Ks + r * 64 + seg * 8) = *(const uint4*)(qp + (size_t)(t0 + r) * DHEAD + seg * 8);
    }
    __syncthreads();
    v16h qa0 = lds_frag_a(Ks + wave * 16 * 64, 64, lane, 0);
    v16h qa1 = lds_frag_a(Ks + wave * 16 * 64, 64, lane, 32);

    v8f o[4] = {};
    float mrow[8], lrow[8];
#pragma unroll
    for (int r = 0; r < 8; ++r) { mrow[r] = -1e30f; lrow[r] = 0.f; }

    for (int st = 0; st <= tq; ++st) {
        int s0 = st * 64;
        __syncthreads();
        for (int i = tid; i < 512; i += 128) {           // K tile, row-major
            int r = i >> 3, seg = i & 7;
            *(uint4*)(Ks + r * 64 + seg * 8) = *(const uint4*)(kp + (size_t)(s0 + r) * DHEAD + seg * 8);
        }
        for (int i = tid; i < 512; i += 128) {           // V tile, transposed into [d][s]
            int s = i >> 3, seg = i & 7;
            uint4 tmp = *(const uint4*)(vp + (size_t)(s0 + s) * DHEAD + seg * 8);
            const _Float16* t8 = (const _Float16*)&tmp;
#pragma unroll
            for (int j = 0; j < 8; ++j) Vs[(seg * 8 + j) * 64 + s] = t8[j];
        }
        __syncthreads();

        // S = q @ k^T  (16x64 per wave, f32)
        v8f sacc[4] = {};
#pragma unroll
        for (int n = 0; n < 4; ++n) {
            v16h b0 = lds_frag_b(Ks + n * 16 * 64, 64, lane, 0);
            sacc[n] = __builtin_amdgcn_wmma_f32_16x16x32_f16(false, qa0, false, b0,
                                                             (short)0, sacc[n], false, false);
            v16h b1 = lds_frag_b(Ks + n * 16 * 64, 64, lane, 32);
            sacc[n] = __builtin_amdgcn_wmma_f32_16x16x32_f16(false, qa1, false, b1,
                                                             (short)0, sacc[n], false, false);
        }

        if (st == tq) {                                   // causal mask on diagonal tile
#pragma unroll
            for (int n = 0; n < 4; ++n)
#pragma unroll
                for (int r = 0; r < 8; ++r) {
                    int trow = t0 + wave * 16 + mbase + r;
                    int scol = s0 + n * 16 + col;
                    if (scol > trow) sacc[n][r] = -1e30f;
                }
        }

        // online softmax: row max / rescale / exp / row sum
        float mnew[8];
#pragma unroll
        for (int r = 0; r < 8; ++r) {
            float mx = mrow[r];
#pragma unroll
            for (int n = 0; n < 4; ++n) mx = fmaxf(mx, sacc[n][r]);
#pragma unroll
            for (int off = 1; off < 16; off <<= 1) mx = fmaxf(mx, __shfl_xor(mx, off, 32));
            mnew[r] = mx;
        }
#pragma unroll
        for (int r = 0; r < 8; ++r) {
            float alpha = __expf(mrow[r] - mnew[r]);
            mrow[r] = mnew[r];
            float s = 0.f;
#pragma unroll
            for (int n = 0; n < 4; ++n) {
                float p = __expf(sacc[n][r] - mnew[r]);
                sacc[n][r] = p;
                s += p;
            }
#pragma unroll
            for (int off = 1; off < 16; off <<= 1) s += __shfl_xor(s, off, 32);
            lrow[r] = lrow[r] * alpha + s;
#pragma unroll
            for (int n = 0; n < 4; ++n) o[n][r] = o[n][r] * alpha;
        }

        // P (f16) -> per-wave LDS strip in [row][s] layout
#pragma unroll
        for (int n = 0; n < 4; ++n)
#pragma unroll
            for (int r = 0; r < 8; ++r)
                Ps[wave * 16 * 64 + (mbase + r) * 64 + n * 16 + col] = (_Float16)sacc[n][r];
        __syncthreads();

        // O += P @ V
#pragma unroll
        for (int kk = 0; kk < 2; ++kk) {
            v16h pa = lds_frag_a(Ps + wave * 16 * 64, 64, lane, kk * 32);
#pragma unroll
            for (int n = 0; n < 4; ++n) {
                v16h vb = lds_frag_b(Vs + n * 16 * 64, 64, lane, kk * 32);
                o[n] = __builtin_amdgcn_wmma_f32_16x16x32_f16(false, pa, false, vb,
                                                              (short)0, o[n], false, false);
            }
        }
    }

    // normalize and write concat-head output Y[b*T+t][h*64+d]
#pragma unroll
    for (int r = 0; r < 8; ++r) {
        float inv = 1.0f / lrow[r];
        int t = t0 + wave * 16 + mbase + r;
#pragma unroll
        for (int n = 0; n < 4; ++n)
            Y[(size_t)(b * SEQ + t) * CDIM + h * DHEAD + n * 16 + col] = (_Float16)(o[n][r] * inv);
    }
}

// ---------------------------------------------------------------------------
// Output projection: out[B*T, C] = Y[B*T, 1024] @ Wp[1024, C] + bp, via WpT[C,1024].
// Same 128x64 double-buffered pipeline as qkv_gemm.
// ---------------------------------------------------------------------------
__global__ __launch_bounds__(128)
void proj_gemm(const _Float16* __restrict__ Y,     // [B*T, 1024]
               const _Float16* __restrict__ WpT,   // [C, 1024]
               const float* __restrict__ bp,
               float* __restrict__ out) {          // [B*T, C]
    int tileM = blockIdx.x;                        // (B*T)/128
    int tileN = blockIdx.y;                        // C/64
    const _Float16* Arow = Y   + (size_t)tileM * 128 * CDIM;
    const _Float16* Brow = WpT + (size_t)tileN * 64  * CDIM;

    __shared__ __align__(16) _Float16 As[2][128 * 32];
    __shared__ __align__(16) _Float16 Bs[2][64 * 32];

    int tid = threadIdx.x, wave = tid >> 5, lane = tid & 31;
    v8f acc[2][4] = {};

    auto stage = [&](int buf, int kc) {
        for (int i = tid; i < 512; i += 128) {
            int r = i >> 2, seg = i & 3;
            copy16_g2l(Arow + (size_t)r * CDIM + kc + seg * 8, &As[buf][r * 32 + seg * 8]);
        }
        for (int i = tid; i < 256; i += 128) {
            int r = i >> 2, seg = i & 3;
            copy16_g2l(Brow + (size_t)r * CDIM + kc + seg * 8, &Bs[buf][r * 32 + seg * 8]);
        }
    };

    stage(0, 0);
    for (int kc = 0, it = 0; kc < CDIM; kc += 32, ++it) {
        int cur = it & 1;
        wait_stage();
        __syncthreads();
        if (kc + 32 < CDIM) stage(cur ^ 1, kc + 32);
        v16h a0 = lds_frag_a(&As[cur][(wave * 32 + 0)  * 32], 32, lane, 0);
        v16h a1 = lds_frag_a(&As[cur][(wave * 32 + 16) * 32], 32, lane, 0);
#pragma unroll
        for (int n = 0; n < 4; ++n) {
            v16h bf = lds_frag_b(&Bs[cur][n * 16 * 32], 32, lane, 0);
            acc[0][n] = __builtin_amdgcn_wmma_f32_16x16x32_f16(false, a0, false, bf,
                                                               (short)0, acc[0][n], false, false);
            acc[1][n] = __builtin_amdgcn_wmma_f32_16x16x32_f16(false, a1, false, bf,
                                                               (short)0, acc[1][n], false, false);
        }
    }

    int mbase = (lane < 16) ? 0 : 8;
    int col = lane & 15;
#pragma unroll
    for (int sub = 0; sub < 2; ++sub)
#pragma unroll
        for (int n = 0; n < 4; ++n) {
            float bias = bp[tileN * 64 + n * 16 + col];
#pragma unroll
            for (int r = 0; r < 8; ++r) {
                int m = tileM * 128 + wave * 32 + sub * 16 + mbase + r;
                out[(size_t)m * CDIM + tileN * 64 + n * 16 + col] = acc[sub][n][r] + bias;
            }
        }
}

// ---------------------------------------------------------------------------
extern "C" void kernel_launch(void* const* d_in, const int* in_sizes, int n_in,
                              void* d_out, int out_size, void* d_ws, size_t ws_size,
                              hipStream_t stream) {
    (void)in_sizes; (void)n_in; (void)out_size; (void)ws_size;
    const float* x  = (const float*)d_in[0];
    const float* Wq = (const float*)d_in[1];
    const float* Wk = (const float*)d_in[2];
    const float* Wv = (const float*)d_in[3];
    const float* Wp = (const float*)d_in[4];
    const float* bp = (const float*)d_in[5];
    float* out = (float*)d_out;

    char* ws = (char*)d_ws;
    size_t off = 0;
    auto alloc = [&](size_t elems) {
        _Float16* p = (_Float16*)(ws + off);
        off += ((elems * sizeof(_Float16) + 255) / 256) * 256;
        return p;
    };
    _Float16* xh  = alloc((size_t)BATCH * SEQ * CDIM);
    _Float16* WqT = alloc((size_t)NHEAD * DHEAD * CDIM);
    _Float16* WkT = alloc((size_t)NHEAD * DHEAD * CDIM);
    _Float16* WvT = alloc((size_t)NHEAD * DHEAD * CDIM);
    _Float16* WpT = alloc((size_t)CDIM * CDIM);
    _Float16* qb  = alloc((size_t)BATCH * NHEAD * SEQ * DHEAD);
    _Float16* kb  = alloc((size_t)BATCH * NHEAD * SEQ * DHEAD);
    _Float16* vb  = alloc((size_t)BATCH * NHEAD * SEQ * DHEAD);
    _Float16* Yb  = alloc((size_t)BATCH * SEQ * CDIM);

    int nx = BATCH * SEQ * CDIM;
    cvt_f32_f16<<<(nx + 255) / 256, 256, 0, stream>>>(x, xh, nx);

    int nw = NHEAD * CDIM * DHEAD;
    transpose_cvt<<<(nw + 255) / 256, 256, 0, stream>>>(Wq, WqT, NHEAD, CDIM, DHEAD);
    transpose_cvt<<<(nw + 255) / 256, 256, 0, stream>>>(Wk, WkT, NHEAD, CDIM, DHEAD);
    transpose_cvt<<<(nw + 255) / 256, 256, 0, stream>>>(Wv, WvT, NHEAD, CDIM, DHEAD);
    int np = CDIM * CDIM;
    transpose_cvt<<<(np + 255) / 256, 256, 0, stream>>>(Wp, WpT, 1, CDIM, CDIM);

    dim3 gq(SEQ / 128, BATCH * NHEAD, 3);
    qkv_gemm<<<gq, 128, 0, stream>>>(xh, WqT, WkT, WvT, qb, kb, vb);

    dim3 ga(SEQ / 64, BATCH * NHEAD);
    attn_kernel<<<ga, 128, 0, stream>>>(qb, kb, vb, Yb);

    dim3 gp((BATCH * SEQ) / 128, CDIM / 64);
    proj_gemm<<<gp, 128, 0, stream>>>(Yb, WpT, bp, out);
}
